// LuongDotProductAttention_45698452030201
// MI455X (gfx1250) — matrix operand order repaired and checked by reading, not compiled, not run on hardware
//
#include <hip/hip_runtime.h>
#include <math.h>

typedef __attribute__((ext_vector_type(16))) _Float16     v16h;
typedef __attribute__((ext_vector_type(8)))  float        v8f;
typedef __attribute__((ext_vector_type(4)))  unsigned int v4u;
typedef __attribute__((ext_vector_type(8)))  int          v8i;
typedef __attribute__((ext_vector_type(4)))  int          v4i;

#define B_ 32
#define S_ 8192
#define D_ 1024
#define NEG_INF_ -1.0e9f

// ---------------------------------------------------------------------------
// Kernel 1: scores[b,s] = enc[b,s,:] . q[b,:]  (masked), via WMMA f16->f32
// grid = B * (S/128), block = 256 (8 waves, 16 rows/wave)
// ---------------------------------------------------------------------------
__global__ __launch_bounds__(256) void luong_scores_kernel(
    const float* __restrict__ q, const float* __restrict__ enc,
    const int* __restrict__ mask, float* __restrict__ scores)
{
    __shared__ _Float16 qsh[D_];

    const int b    = blockIdx.x >> 6;   // 64 row-tiles of 128 per batch
    const int tile = blockIdx.x & 63;
    const int tid  = threadIdx.x;

    // Stage q[b,:] into LDS as f16 (coalesced float4 loads).
    {
        const float4 f = *(const float4*)(q + (size_t)b * D_ + tid * 4);
        qsh[tid * 4 + 0] = (_Float16)f.x;
        qsh[tid * 4 + 1] = (_Float16)f.y;
        qsh[tid * 4 + 2] = (_Float16)f.z;
        qsh[tid * 4 + 3] = (_Float16)f.w;
    }
    __syncthreads();

    const int wave = tid >> 5;
    const int lane = tid & 31;
    const int m    = lane & 15;   // A-matrix row within the wave's 16-row tile
    const int h    = lane >> 4;   // lane-half selector

    const int    srow = tile * 128 + wave * 16 + m;
    const float* arow = enc + ((size_t)b * S_ + srow) * D_;

    v8f acc = {};
    for (int d0 = 0; d0 < D_; d0 += 32) {
        // A layout (16-bit A 16x32): lane half h holds K = h*8+{0..7}, h*8+16+{0..7}
        const float4 f0 = *(const float4*)(arow + d0 + h * 8);
        const float4 f1 = *(const float4*)(arow + d0 + h * 8 + 4);
        const float4 f2 = *(const float4*)(arow + d0 + h * 8 + 16);
        const float4 f3 = *(const float4*)(arow + d0 + h * 8 + 20);
        v16h a;
        a[0]  = (_Float16)f0.x; a[1]  = (_Float16)f0.y;
        a[2]  = (_Float16)f0.z; a[3]  = (_Float16)f0.w;
        a[4]  = (_Float16)f1.x; a[5]  = (_Float16)f1.y;
        a[6]  = (_Float16)f1.z; a[7]  = (_Float16)f1.w;
        a[8]  = (_Float16)f2.x; a[9]  = (_Float16)f2.y;
        a[10] = (_Float16)f2.z; a[11] = (_Float16)f2.w;
        a[12] = (_Float16)f3.x; a[13] = (_Float16)f3.y;
        a[14] = (_Float16)f3.z; a[15] = (_Float16)f3.w;

        // B layout (32x16): lane half h holds K = h*8+{0..7}, h*8+16+{0..7}; replicate q
        // across all 16 columns so every output column equals the score.
        v16h bv;
        #pragma unroll
        for (int i = 0; i < 16; ++i) bv[i] = qsh[d0 + h * 8 + i + (i >= 8 ? 8 : 0)];

        acc = __builtin_amdgcn_wmma_f32_16x16x32_f16(
            false, a, false, bv, (short)0, acc, false, false);
    }

    // C layout: lane half h, VGPR r holds row M = h*8 + r (all columns equal).
    if (m == 0) {
        const int rbase = tile * 128 + wave * 16 + h * 8;
        #pragma unroll
        for (int r = 0; r < 8; ++r) {
            const int row = rbase + r;
            const float sc = mask[(size_t)b * S_ + row] ? acc[r] : NEG_INF_;
            scores[(size_t)b * S_ + row] = sc;
        }
    }
}

// ---------------------------------------------------------------------------
// Kernel 2: in-place softmax over S per batch. grid = B, block = 256.
// ---------------------------------------------------------------------------
__global__ __launch_bounds__(256) void luong_softmax_kernel(float* __restrict__ wts)
{
    const int b = blockIdx.x;
    const int t = threadIdx.x;
    float* sc = wts + (size_t)b * S_;

    float4 v[8];
    float mx = -INFINITY;
    #pragma unroll
    for (int j = 0; j < 8; ++j) {
        v[j] = *(const float4*)(sc + (j * 256 + t) * 4);
        mx = fmaxf(mx, fmaxf(fmaxf(v[j].x, v[j].y), fmaxf(v[j].z, v[j].w)));
    }

    __shared__ float red[256];
    red[t] = mx;
    __syncthreads();
    for (int s = 128; s > 0; s >>= 1) {
        if (t < s) red[t] = fmaxf(red[t], red[t + s]);
        __syncthreads();
    }
    mx = red[0];
    __syncthreads();

    float sum = 0.0f;
    #pragma unroll
    for (int j = 0; j < 8; ++j) {
        v[j].x = __expf(v[j].x - mx);
        v[j].y = __expf(v[j].y - mx);
        v[j].z = __expf(v[j].z - mx);
        v[j].w = __expf(v[j].w - mx);
        sum += v[j].x + v[j].y + v[j].z + v[j].w;
    }
    red[t] = sum;
    __syncthreads();
    for (int s = 128; s > 0; s >>= 1) {
        if (t < s) red[t] += red[t + s];
        __syncthreads();
    }
    const float inv = 1.0f / red[0];

    #pragma unroll
    for (int j = 0; j < 8; ++j) {
        float4 o;
        o.x = v[j].x * inv; o.y = v[j].y * inv;
        o.z = v[j].z * inv; o.w = v[j].w * inv;
        *(float4*)(sc + (j * 256 + t) * 4) = o;
    }
}

// ---------------------------------------------------------------------------
// Kernel 3: context[b,d] = sum_s w[b,s] * enc[b,s,d], via WMMA.
// grid = B * (D/128), block = 256 (8 waves, 16 d-columns per wave).
// TDM (tensor_load_to_lds) double-buffers the 32x128 f32 enc tile into LDS
// with a 1-DWORD row pad (129-DWORD stride) for conflict-free column reads.
// ---------------------------------------------------------------------------
#if __has_builtin(__builtin_amdgcn_tensor_load_to_lds)
#define USE_TDM 1
#else
#define USE_TDM 0
#endif

#if USE_TDM
// Issue a TDM load of a 32(s) x 128(d) f32 tile starting at gptr into LDS
// at byte offset lds_off. Row stride in memory = 1024 floats; LDS rows are
// padded by 1 DWORD every 128 DWORDs (pad_interval=6, pad_amount=0).
__device__ __forceinline__ void tdm_issue_tile(const float* gptr, unsigned lds_off)
{
    const unsigned long long ga = (unsigned long long)(uintptr_t)gptr;
    // D# group 0: count=1 | lds_addr | global_addr[56:0] | type=2
    v4u g0 = { 1u,
               lds_off,
               (unsigned)(ga & 0xffffffffull),
               (unsigned)((ga >> 32) & 0x01ffffffull) | (2u << 30) };
    // D# group 1:
    //  w0: data_size=2(4B)<<16 | pad_enable<<20 | pad_interval=6<<22 (128 DW) | pad_amount=0 (1 DW)
    //  w1: atomic_barrier_addr=0 | tensor_dim0[15:0]=1024 <<16
    //  w2: tensor_dim0[31:16]=0 | tensor_dim1[15:0]=8192 <<16
    //  w3: tensor_dim1[31:16]=0 | tile_dim0=128 <<16
    //  w4: tile_dim1=32 | tile_dim2=0
    //  w5: tensor_dim0_stride[31:0]=1024   w6,w7: 0
    v8i g1 = { (int)((2u << 16) | (1u << 20) | (6u << 22)),
               (int)(1024u << 16),
               (int)(8192u << 16),
               (int)(128u << 16),
               32, 1024, 0, 0 };
    v4i gz4 = { 0, 0, 0, 0 };           // groups 2/3 unused (<=2D tensor)
    v8i gz8 = { 0, 0, 0, 0, 0, 0, 0, 0 };
    // 6-arg toolchain form (clang-23 / therock-10.0 headers).
    __builtin_amdgcn_tensor_load_to_lds(g0, g1, gz4, gz4, gz8, 0);
}
#endif

__global__ __launch_bounds__(256) void luong_context_kernel(
    const float* __restrict__ enc, const float* __restrict__ wts,
    float* __restrict__ ctx)
{
#if USE_TDM
    __shared__ float    etile[2][32][129];   // 129-DWORD stride matches TDM pad
    __shared__ _Float16 wtile[32];

    const int b    = blockIdx.x >> 3;        // 8 column-blocks of 128 per batch
    const int dblk = (blockIdx.x & 7) * 128;
    const int tid  = threadIdx.x;
    const int wave = tid >> 5;
    const int lane = tid & 31;
    const int n    = lane & 15;
    const int h    = lane >> 4;

    const float* wp    = wts + (size_t)b * S_;
    const float* tbase = enc + ((size_t)b * S_) * D_ + dblk;
    const int    NIT   = S_ / 32;            // 256 tiles

    if (wave == 0)
        tdm_issue_tile(tbase, (unsigned)(uintptr_t)&etile[0][0][0]);

    v8f acc = {};
    for (int it = 0; it < NIT; ++it) {
        if (wave == 0) {
            if (it + 1 < NIT) {
                tdm_issue_tile(tbase + (size_t)(it + 1) * 32 * D_,
                               (unsigned)(uintptr_t)&etile[(it + 1) & 1][0][0]);
                __builtin_amdgcn_s_wait_tensorcnt(1);   // tile `it` resident
            } else {
                __builtin_amdgcn_s_wait_tensorcnt(0);
            }
        }
        if (wave == 1)
            wtile[lane] = (_Float16)wp[it * 32 + lane];
        __syncthreads();

        const float (*et)[129] = etile[it & 1];

        // A: weights chunk replicated over all 16 rows (A K-pattern).
        v16h a;
        #pragma unroll
        for (int i = 0; i < 8; ++i)  a[i] = wtile[h * 16 + i];
        #pragma unroll
        for (int i = 8; i < 16; ++i) a[i] = wtile[h * 16 + i];

        // B: enc tile column (wave*16+n), K = h*16 + i (s-direction), cvt f32->f16.
        v16h bv;
        #pragma unroll
        for (int i = 0; i < 16; ++i)
            bv[i] = (_Float16)et[h * 16 + i][wave * 16 + n];

        acc = __builtin_amdgcn_wmma_f32_16x16x32_f16(
            false, a, false, bv, (short)0, acc, false, false);
        __syncthreads();
    }

    if (h == 0)
        ctx[(size_t)b * D_ + dblk + wave * 16 + n] = acc[0];
#else
    // Fallback: manual coalesced staging (f32 -> f16) without TDM.
    __shared__ _Float16 etile[32][136];
    __shared__ _Float16 wtile[32];

    const int b    = blockIdx.x >> 3;
    const int dblk = (blockIdx.x & 7) * 128;
    const int tid  = threadIdx.x;
    const int wave = tid >> 5;
    const int lane = tid & 31;
    const int n    = lane & 15;
    const int h    = lane >> 4;

    const float* wp = wts + (size_t)b * S_;

    v8f acc = {};
    for (int s0 = 0; s0 < S_; s0 += 32) {
        __syncthreads();
        #pragma unroll
        for (int j = 0; j < 4; ++j) {
            const int idx = j * 256 + tid;
            const int row = idx >> 5;
            const int c4  = (idx & 31) * 4;
            const float4 f = *(const float4*)(enc +
                ((size_t)b * S_ + s0 + row) * D_ + dblk + c4);
            etile[row][c4 + 0] = (_Float16)f.x;
            etile[row][c4 + 1] = (_Float16)f.y;
            etile[row][c4 + 2] = (_Float16)f.z;
            etile[row][c4 + 3] = (_Float16)f.w;
        }
        if (tid < 32) wtile[tid] = (_Float16)wp[s0 + tid];
        __syncthreads();

        v16h a;
        #pragma unroll
        for (int i = 0; i < 8; ++i)  a[i] = wtile[h * 16 + i];
        #pragma unroll
        for (int i = 8; i < 16; ++i) a[i] = wtile[h * 16 + i];
        v16h bv;
        #pragma unroll
        for (int i = 0; i < 16; ++i) bv[i] = etile[h * 16 + i][wave * 16 + n];

        acc = __builtin_amdgcn_wmma_f32_16x16x32_f16(
            false, a, false, bv, (short)0, acc, false, false);
    }
    if (h == 0)
        ctx[(size_t)b * D_ + dblk + wave * 16 + n] = acc[0];
#endif
}

// ---------------------------------------------------------------------------
extern "C" void kernel_launch(void* const* d_in, const int* in_sizes, int n_in,
                              void* d_out, int out_size, void* d_ws, size_t ws_size,
                              hipStream_t stream) {
    (void)in_sizes; (void)n_in; (void)out_size; (void)d_ws; (void)ws_size;
    const float* q    = (const float*)d_in[0];
    const float* enc  = (const float*)d_in[1];
    const int*   mask = (const int*)d_in[2];

    float* ctx = (float*)d_out;            // [B, D]
    float* wts = (float*)d_out + B_ * D_;  // [B, S]; scores then weights in place

    luong_scores_kernel<<<B_ * (S_ / 128), 256, 0, stream>>>(q, enc, mask, wts);
    luong_softmax_kernel<<<B_, 256, 0, stream>>>(wts);
    luong_context_kernel<<<B_ * (D_ / 128), 256, 0, stream>>>(enc, wts, ctx);
}
